// Decoder_89893665506099
// MI455X (gfx1250) — matrix operand order repaired and checked by reading, not compiled, not run on hardware
//
#include <hip/hip_runtime.h>
#include <hip/hip_bf16.h>

typedef __attribute__((ext_vector_type(16))) _Float16 v16h;
typedef __attribute__((ext_vector_type(8)))  _Float16 v8h;
typedef __attribute__((ext_vector_type(8)))  float    v8f;
typedef __attribute__((ext_vector_type(4)))  float    v4f;

#define B_   512
#define T_   80
#define DENC 1024
#define DF   512
#define DW   512
#define H_   512
#define G4   2048   // 4*H

// ---------------------------------------------------------------- WMMA helpers

__device__ __forceinline__ v8f wmma_f16(v16h a, v16h b, v8f c) {
  // v_wmma_f32_16x16x32_f16 : D = A(16x32 f16) x B(32x16 f16) + C(16x16 f32)
  return __builtin_amdgcn_wmma_f32_16x16x32_f16(false, a, false, b, (short)0, c,
                                                false, false);
}

// A fragment (16x32, f16 row-major source). a_row = this lane's row start.
// Lane l<16 holds row m=l, K = {kb+0..7, kb+16..23}; lane>=16 holds row m=l-16,
// K = {kb+8..15, kb+24..31}  (per CDNA5 ISA 16-bit A layout).
__device__ __forceinline__ v16h afrag_f16(const _Float16* a_row, int kb, int lane) {
  const int koff = (lane >> 4) << 3;
  v8h g0 = *(const v8h*)(a_row + kb + koff);
  v8h g1 = *(const v8h*)(a_row + kb + 16 + koff);
  v16h r;
#pragma unroll
  for (int i = 0; i < 8; ++i) { r[i] = g0[i]; r[i + 8] = g1[i]; }
  return r;
}

// A fragment from f32 row-major source, converted to f16 in registers.
__device__ __forceinline__ v16h afrag_f32(const float* a_row, int kb, int lane) {
  const int koff = (lane >> 4) << 3;
  v4f g0 = *(const v4f*)(a_row + kb + koff);
  v4f g1 = *(const v4f*)(a_row + kb + koff + 4);
  v4f g2 = *(const v4f*)(a_row + kb + 16 + koff);
  v4f g3 = *(const v4f*)(a_row + kb + 16 + koff + 4);
  v16h r;
#pragma unroll
  for (int i = 0; i < 4; ++i) {
    r[i]      = (_Float16)g0[i];
    r[i + 4]  = (_Float16)g1[i];
    r[i + 8]  = (_Float16)g2[i];
    r[i + 12] = (_Float16)g3[i];
  }
  return r;
}

// B fragment: weights stored [N, K] f16 row-major; B = Wᵀ so column n of B is
// row n of W -> two 8-half runs per lane. w_row = this lane's W row.
// Lane l<16: K = {kb+0..7, kb+16..23} ; lane>=16: K = {kb+8..15, kb+24..31}.
__device__ __forceinline__ v16h bfrag_w(const _Float16* w_row, int kb, int lane) {
  const int koff = (lane >> 4) << 3;
  v8h g0 = *(const v8h*)(w_row + kb + koff);
  v8h g1 = *(const v8h*)(w_row + kb + 16 + koff);
  v16h r;
#pragma unroll
  for (int i = 0; i < 8; ++i) { r[i] = g0[i]; r[i + 8] = g1[i]; }
  return r;
}

__device__ __forceinline__ float fsigmoid(float x) {
  return 1.f / (1.f + __expf(-x));
}
__device__ __forceinline__ float ftanh_(float x) {
  return 2.f / (1.f + __expf(-2.f * x)) - 1.f;
}

// ---------------------------------------------------------------- prep kernel
// One-shot weight conversion / layout change; grid = G4*DF threads.
__global__ void lstm_prep_kernel(const float* __restrict__ W_proj,
                                 const float* __restrict__ embed,
                                 const int*   __restrict__ start_ids,
                                 const float* __restrict__ W_ih,
                                 const float* __restrict__ W_hh,
                                 const float* __restrict__ b_ih,
                                 const float* __restrict__ b_hh,
                                 _Float16* __restrict__ Wp_t,    // [DF, DENC]
                                 _Float16* __restrict__ Wih_f,   // [G4, DF]
                                 _Float16* __restrict__ Wih_w,   // [G4, DW]
                                 _Float16* __restrict__ Whh,     // [G4, H]
                                 _Float16* __restrict__ word16,  // [B, DW]
                                 float*    __restrict__ bias,    // [G4]
                                 _Float16* __restrict__ h16,     // [B, H]
                                 float*    __restrict__ c_st) {  // [B, H]
  const int idx = blockIdx.x * blockDim.x + threadIdx.x;  // < G4*DF
  {
    const int n = idx >> 9, k = idx & 511;
    Wih_f[idx] = (_Float16)W_ih[n * (DF + DW) + k];
    Wih_w[idx] = (_Float16)W_ih[n * (DF + DW) + DF + k];
    Whh[idx]   = (_Float16)W_hh[idx];
  }
  if (idx < DF * DENC) {            // transposed projection weights
    const int n = idx >> 10, k = idx & 1023;
    Wp_t[idx] = (_Float16)W_proj[k * DF + n];
  }
  if (idx < B_ * DW) {              // start-token embedding gather + state init
    const int b = idx >> 9, k = idx & 511;
    word16[idx] = (_Float16)embed[(long)start_ids[b] * DW + k];
    h16[idx]    = (_Float16)0.f;
    c_st[idx]   = 0.f;
  }
  if (idx < G4) bias[idx] = b_ih[idx] + b_hh[idx];
}

// ------------------------------------------------------- projection GEMM
// feats[b*T+t, :] = enc[b,t,:] @ W_proj + b_proj, output stored f16.
// One wave per 16x64 strip: one A fragment feeds 4 WMMAs. All fragments of a
// K-chunk are materialized before the WMMA burst so their loads overlap MACs.
__global__ void __launch_bounds__(128)
proj_gemm_kernel(const float* __restrict__ enc,       // [B*T, DENC]
                 const _Float16* __restrict__ Wp_t,   // [DF, DENC]
                 const float* __restrict__ b_proj,    // [DF]
                 _Float16* __restrict__ feats) {      // [B*T, DF]
  const int lane = threadIdx.x & 31;
  const int tile = blockIdx.x * 4 + (threadIdx.x >> 5);  // (B*T/16)*(DF/64)
  const int n_grp  = tile & 7;         // DF/64 = 8 strips
  const int m_tile = tile >> 3;
  const long m0 = (long)m_tile * 16;
  const int  n0 = n_grp * 64;

  const float* a_row = enc + (m0 + (lane & 15)) * (long)DENC;
  const _Float16* w_row[4];
#pragma unroll
  for (int j = 0; j < 4; ++j)
    w_row[j] = Wp_t + (long)(n0 + 16 * j + (lane & 15)) * DENC;

  v8f acc[4] = {};
#pragma unroll 2
  for (int kb = 0; kb < DENC; kb += 32) {
    v16h bb[4];
#pragma unroll
    for (int j = 0; j < 4; ++j) bb[j] = bfrag_w(w_row[j], kb, lane);
    const v16h a = afrag_f32(a_row, kb, lane);
#pragma unroll
    for (int j = 0; j < 4; ++j) acc[j] = wmma_f16(a, bb[j], acc[j]);
  }

  const int mhi = (lane >> 4) * 8;
#pragma unroll
  for (int j = 0; j < 4; ++j) {
    const int   n  = n0 + 16 * j + (lane & 15);
    const float bn = b_proj[n];
#pragma unroll
    for (int i = 0; i < 8; ++i)
      feats[(m0 + mhi + i) * DF + n] = (_Float16)(acc[j][i] + bn);
  }
}

// ------------------------------------------------- word contribution GEMM
// wordc[b, n] = word16[b,:] @ W_ih[:, DF:]ᵀ[:,n] + (b_ih+b_hh)[n]  (t-invariant)
// One wave per 16x64 strip.
__global__ void __launch_bounds__(128)
wordc_gemm_kernel(const _Float16* __restrict__ word16,  // [B, DW]
                  const _Float16* __restrict__ Wih_w,   // [G4, DW]
                  const float* __restrict__ bias,       // [G4]
                  float* __restrict__ wordc) {          // [B, G4]
  const int lane = threadIdx.x & 31;
  const int tile = blockIdx.x * 4 + (threadIdx.x >> 5);  // 32 m-tiles x 32 strips
  const int n_grp  = tile & 31;        // G4/64 = 32 strips
  const int m_tile = tile >> 5;
  const int m0 = m_tile * 16, n0 = n_grp * 64;

  const _Float16* a_row = word16 + (long)(m0 + (lane & 15)) * DW;
  const _Float16* w_row[4];
#pragma unroll
  for (int j = 0; j < 4; ++j)
    w_row[j] = Wih_w + (long)(n0 + 16 * j + (lane & 15)) * DW;

  v8f acc[4] = {};
#pragma unroll 2
  for (int kb = 0; kb < DW; kb += 32) {
    v16h bb[4];
#pragma unroll
    for (int j = 0; j < 4; ++j) bb[j] = bfrag_w(w_row[j], kb, lane);
    const v16h a = afrag_f16(a_row, kb, lane);
#pragma unroll
    for (int j = 0; j < 4; ++j) acc[j] = wmma_f16(a, bb[j], acc[j]);
  }

  const int mhi = (lane >> 4) * 8;
#pragma unroll
  for (int j = 0; j < 4; ++j) {
    const int   n  = n0 + 16 * j + (lane & 15);
    const float bn = bias[n];
#pragma unroll
    for (int i = 0; i < 8; ++i)
      wordc[(long)(m0 + mhi + i) * G4 + n] = acc[j][i] + bn;
  }
}

// ----------------------------------------------------------- LSTM step
// Block = 4 waves = gates {i,f,g,o} of one 16(batch) x 32(hidden) tile.
// Gate tile = feats_t @ Wih_fᵀ + h @ Whhᵀ  (fused K=1024 WMMA loop, one A
// fragment feeds 2 WMMAs), then block-local elementwise LSTM update via LDS.
__global__ void __launch_bounds__(128)
lstm_step_kernel(const _Float16* __restrict__ feats,   // [B*T, DF]
                 const _Float16* __restrict__ h_in,    // [B, H] f16 (read)
                 const _Float16* __restrict__ Wih_f,   // [G4, DF]
                 const _Float16* __restrict__ Whh,     // [G4, H]
                 const float* __restrict__ wordc,      // [B, G4] (bias folded)
                 float* __restrict__ c_st,             // [B, H]
                 _Float16* __restrict__ h_out,         // [B, H] f16 (write)
                 float* __restrict__ out,              // [B, T, H]
                 int t) {
  __shared__ float gbuf[4][16][32];

  const int lane = threadIdx.x & 31;
  const int gate = threadIdx.x >> 5;          // wave id == gate id
  const int b0 = blockIdx.x * 16;             // batch tile
  const int j0 = blockIdx.y * 32;             // hidden strip (2 x 16)

  const _Float16* wf_row[2];
  const _Float16* wh_row[2];
#pragma unroll
  for (int s = 0; s < 2; ++s) {
    const int nrow = gate * H_ + j0 + 16 * s + (lane & 15);  // gate row of W
    wf_row[s] = Wih_f + (long)nrow * DF;
    wh_row[s] = Whh   + (long)nrow * H_;
  }
  const _Float16* af_row = feats + ((long)(b0 + (lane & 15)) * T_ + t) * DF;
  const _Float16* ah_row = h_in  + (long)(b0 + (lane & 15)) * H_;

  v8f acc[2] = {};
#pragma unroll 2
  for (int kb = 0; kb < DF; kb += 32) {
    v16h bb[2];
#pragma unroll
    for (int s = 0; s < 2; ++s) bb[s] = bfrag_w(wf_row[s], kb, lane);
    const v16h a = afrag_f16(af_row, kb, lane);
#pragma unroll
    for (int s = 0; s < 2; ++s) acc[s] = wmma_f16(a, bb[s], acc[s]);
  }
#pragma unroll 2
  for (int kb = 0; kb < H_; kb += 32) {
    v16h bb[2];
#pragma unroll
    for (int s = 0; s < 2; ++s) bb[s] = bfrag_w(wh_row[s], kb, lane);
    const v16h a = afrag_f16(ah_row, kb, lane);
#pragma unroll
    for (int s = 0; s < 2; ++s) acc[s] = wmma_f16(a, bb[s], acc[s]);
  }

  const int n   = lane & 15;
  const int mhi = (lane >> 4) * 8;
#pragma unroll
  for (int s = 0; s < 2; ++s)
#pragma unroll
    for (int i = 0; i < 8; ++i) gbuf[gate][mhi + i][16 * s + n] = acc[s][i];
  __syncthreads();

#pragma unroll
  for (int e2 = 0; e2 < 4; ++e2) {
    const int e = threadIdx.x + e2 * 128;     // 512 elements / 128 threads
    const int m = e >> 5, j = e & 31;
    const int b = b0 + m, jj = j0 + j;
    const long wcb = (long)b * G4 + jj;
    const float gi = gbuf[0][m][j] + wordc[wcb];
    const float gf = gbuf[1][m][j] + wordc[wcb + H_];
    const float gg = gbuf[2][m][j] + wordc[wcb + 2 * H_];
    const float go = gbuf[3][m][j] + wordc[wcb + 3 * H_];
    const long ci = (long)b * H_ + jj;
    const float cn = fsigmoid(gf) * c_st[ci] + fsigmoid(gi) * ftanh_(gg);
    const float hn = fsigmoid(go) * ftanh_(cn);
    c_st[ci] = cn;
    h_out[ci] = (_Float16)hn;
    out[((long)b * T_ + t) * H_ + jj] = hn;
  }
}

// ---------------------------------------------------------------- launcher
extern "C" void kernel_launch(void* const* d_in, const int* in_sizes, int n_in,
                              void* d_out, int out_size, void* d_ws, size_t ws_size,
                              hipStream_t stream) {
  (void)in_sizes; (void)n_in; (void)out_size; (void)ws_size;
  const float* enc    = (const float*)d_in[0];
  const int*   sid    = (const int*)  d_in[1];
  const float* W_proj = (const float*)d_in[2];
  const float* b_proj = (const float*)d_in[3];
  const float* embed  = (const float*)d_in[4];
  const float* W_ih   = (const float*)d_in[5];
  const float* W_hh   = (const float*)d_in[6];
  const float* b_ih   = (const float*)d_in[7];
  const float* b_hh   = (const float*)d_in[8];
  float* out = (float*)d_out;

  char* ws = (char*)d_ws;
  size_t off = 0;
  auto alloc = [&](size_t bytes) -> char* {
    char* p = ws + off;
    off += (bytes + 255) & ~(size_t)255;
    return p;
  };
  _Float16* feats  = (_Float16*)alloc((size_t)B_ * T_ * DF * 2);  // ~42 MB
  _Float16* Wp_t   = (_Float16*)alloc((size_t)DF * DENC * 2);
  _Float16* Wih_f  = (_Float16*)alloc((size_t)G4 * DF * 2);
  _Float16* Wih_w  = (_Float16*)alloc((size_t)G4 * DW * 2);
  _Float16* Whh    = (_Float16*)alloc((size_t)G4 * H_ * 2);
  _Float16* word16 = (_Float16*)alloc((size_t)B_ * DW * 2);
  float*    bias   = (float*)   alloc((size_t)G4 * 4);
  float*    wordc  = (float*)   alloc((size_t)B_ * G4 * 4);
  _Float16* h_a    = (_Float16*)alloc((size_t)B_ * H_ * 2);
  _Float16* h_b    = (_Float16*)alloc((size_t)B_ * H_ * 2);
  float*    c_st   = (float*)   alloc((size_t)B_ * H_ * 4);

  // 1) weight conversion / embedding gather / state init
  lstm_prep_kernel<<<(G4 * DF) / 256, 256, 0, stream>>>(
      W_proj, embed, sid, W_ih, W_hh, b_ih, b_hh,
      Wp_t, Wih_f, Wih_w, Whh, word16, bias, h_a, c_st);

  // 2) encoder projection (big parallel WMMA GEMM), feats stored f16
  proj_gemm_kernel<<<(B_ * T_ / 16) * (DF / 64) / 4, 128, 0, stream>>>(
      enc, Wp_t, b_proj, feats);

  // 3) time-invariant word contribution + biases
  wordc_gemm_kernel<<<(B_ / 16) * (G4 / 64) / 4, 128, 0, stream>>>(
      word16, Wih_w, bias, wordc);

  // 4) 80 recurrent steps; h f16 ping-pong to avoid cross-block RAW races
  for (int t = 0; t < T_; ++t) {
    const _Float16* hin  = (t & 1) ? h_b : h_a;
    _Float16*       hout = (t & 1) ? h_a : h_b;
    lstm_step_kernel<<<dim3(B_ / 16, H_ / 32), 128, 0, stream>>>(
        feats, hin, Wih_f, Whh, wordc, c_st, hout, out, t);
  }
}
